// MultiHeadAttention_274877907013
// MI455X (gfx1250) — matrix operand order repaired and checked by reading, not compiled, not run on hardware
//
#include <hip/hip_runtime.h>

// ---------------------------------------------------------------- types
typedef __attribute__((ext_vector_type(16))) _Float16 v16h;
typedef __attribute__((ext_vector_type(8)))  float    v8f;
typedef __attribute__((ext_vector_type(4)))  int      v4i;

union Frag16 { v16h v; uint4 q[2]; };

#define WMMA_F16(A, B, C) \
    __builtin_amdgcn_wmma_f32_16x16x32_f16(false, (A), false, (B), (short)0, (C), false, false)

// ---------------------------------------------------------------- constants
#define BATCH  2
#define SLEN   2048
#define NH     16
#define HD     64
#define DMODEL 1024

// ---------------------------------------------------------------- async global->LDS copy
#if __has_builtin(__builtin_amdgcn_global_load_async_to_lds_b128)
#define HAVE_ASYNC_LDS 1
#else
#define HAVE_ASYNC_LDS 0
#endif

#define GLOBAL_AS __attribute__((address_space(1)))
#define LDS_AS    __attribute__((address_space(3)))

__device__ inline void async_copy16(_Float16* lds_dst, const _Float16* gsrc) {
#if HAVE_ASYNC_LDS
    __builtin_amdgcn_global_load_async_to_lds_b128(
        (GLOBAL_AS v4i*)gsrc, (LDS_AS v4i*)lds_dst, 0, 0);
#else
    *(uint4*)lds_dst = *(const uint4*)gsrc;
#endif
}

__device__ inline void wait_async0() {
#if HAVE_ASYNC_LDS
#if __has_builtin(__builtin_amdgcn_s_wait_asynccnt)
    __builtin_amdgcn_s_wait_asynccnt(0);
#else
    asm volatile("s_wait_asynccnt 0x0" ::: "memory");
#endif
#endif
}

// ---------------------------------------------------------------- DPP16 half-wave max
template <int CTRL>
__device__ inline float dpp_xchg(float x) {
    int xi = __builtin_bit_cast(int, x);
    int r  = __builtin_amdgcn_update_dpp(xi, xi, CTRL, 0xF, 0xF, false);
    return __builtin_bit_cast(float, r);
}
__device__ inline float rowmax16(float x) {
    x = fmaxf(x, dpp_xchg<0xB1>(x));    // quad_perm [1,0,3,2]
    x = fmaxf(x, dpp_xchg<0x4E>(x));    // quad_perm [2,3,0,1]
    x = fmaxf(x, dpp_xchg<0x141>(x));   // row_half_mirror
    x = fmaxf(x, dpp_xchg<0x140>(x));   // row_mirror
    return x;
}

// Load a 32x16 f16 B-fragment: lane = column n, halves = 16 contiguous k from `col`.
__device__ inline v16h load_bfrag(const _Float16* __restrict__ buf, int row, int col) {
    Frag16 f;
    f.q[0] = *(const uint4*)(buf + row * 64 + col);
    f.q[1] = *(const uint4*)(buf + row * 64 + col + 8);
    return f.v;
}

__device__ inline v16h ones_frag() {
    v16h o;
#pragma unroll
    for (int i = 0; i < 16; ++i) o[i] = (_Float16)1.0f;
    return o;
}

// ---------------------------------------------------------------- fp32 -> f16 convert
__global__ void convert_kernel(const float* __restrict__ src, _Float16* __restrict__ dst,
                               int n, float scale) {
    int i = blockIdx.x * blockDim.x + threadIdx.x;
    int stride = gridDim.x * blockDim.x;
    for (; i < n; i += stride) dst[i] = (_Float16)(src[i] * scale);
}

// ---------------------------------------------------------------- V: fp32 (B,S,H,hd) -> f16 (B,H,hd,S)
__global__ __launch_bounds__(256)
void convert_v_transpose(const float* __restrict__ v, _Float16* __restrict__ vt) {
    __shared__ _Float16 tile[64 * 65];
    const int s0 = blockIdx.x * 64;
    const int bh = blockIdx.y;
    const int b  = bh >> 4;
    const int h  = bh & 15;
    const long rowStride = (long)NH * HD;
    const float* src = v + (long)b * SLEN * rowStride + (long)h * HD;

    for (int c = threadIdx.x; c < 64 * 64; c += 256) {
        int s = c >> 6, d = c & 63;
        tile[d * 65 + s] = (_Float16)src[(long)(s0 + s) * rowStride + d];
    }
    __syncthreads();
    for (int c = threadIdx.x; c < 64 * 64; c += 256) {
        int d = c >> 6, s = c & 63;
        vt[((long)bh * HD + d) * SLEN + s0 + s] = tile[d * 65 + s];
    }
}

// ---------------------------------------------------------------- flash attention
// grid: (16, 32) = (S/128 query tiles, B*H). block: 256 threads = 8 waves.
__global__ __launch_bounds__(256)
void attn_kernel(const _Float16* __restrict__ qh, const _Float16* __restrict__ kh,
                 const _Float16* __restrict__ vt, _Float16* __restrict__ xh) {
    __shared__ _Float16 kbuf[2][64 * 64];    // K block [key][d]
    __shared__ _Float16 vtbuf[2][64 * 64];   // V^T block [d][key]
    __shared__ _Float16 pbuf[8 * 16 * 64];   // per-wave P staging

    const int tid  = threadIdx.x;
    const int lane = tid & 31;
    const int wave = tid >> 5;
    const int n    = lane & 15;
    const int hi   = lane >> 4;

    const int qtile = blockIdx.x;
    const int bh    = blockIdx.y;
    const int b     = bh >> 4;
    const int h     = bh & 15;

    const long rowStride = (long)NH * HD;     // 1024
    const long base  = (long)b * SLEN * rowStride + (long)h * HD;
    const _Float16* vtb = vt + (long)bh * HD * SLEN;

    const int qrow0 = qtile * 128 + wave * 16;
    const int myrow = qrow0 + n;

    // Q A-fragments (2 k-chunks), loaded once
    Frag16 aq[2];
    {
        const _Float16* qr = qh + base + (long)myrow * rowStride;
#pragma unroll
        for (int kk = 0; kk < 2; ++kk) {
            aq[kk].q[0] = *(const uint4*)(qr + kk * 32 + hi * 16);
            aq[kk].q[1] = *(const uint4*)(qr + kk * 32 + 8 + hi * 16);
        }
    }

    v8f o0 = {}, o1 = {}, o2 = {}, o3 = {};
    v8f lacc = {};                 // row sums, accumulated on the matrix core
    float m_r[8];
#pragma unroll
    for (int g = 0; g < 8; ++g) m_r[g] = -1e30f;

    _Float16* myp = pbuf + wave * (16 * 64);
    const v16h vones = ones_frag();

    // Per-thread straight-line staging: chunk c -> LDS halves [8c, 8c+8)
    auto stage = [&](int kb, int bufi) {
        const long kbase = base + (long)(kb * 64) * rowStride;
#pragma unroll
        for (int u = 0; u < 2; ++u) {
            int c = tid + u * 256;
            async_copy16(&kbuf[bufi][c * 8],
                         kh + kbase + (long)(c >> 3) * rowStride + (c & 7) * 8);
        }
        const _Float16* vrow = vtb + kb * 64;
#pragma unroll
        for (int u = 0; u < 2; ++u) {
            int c = tid + u * 256;
            async_copy16(&vtbuf[bufi][c * 8],
                         vrow + (long)(c >> 3) * SLEN + (c & 7) * 8);
        }
    };

    const int NB = SLEN / 64;
    stage(0, 0);
    wait_async0();
    __syncthreads();

    for (int kb = 0; kb < NB; ++kb) {
        const int cur = kb & 1;
        if (kb + 1 < NB) stage(kb + 1, cur ^ 1);   // async prefetch next block

        const _Float16* kcur = kbuf[cur];
        const _Float16* vcur = vtbuf[cur];

        // ---- S = Q * K^T (scale folded into Q)
        v8f s0 = {}, s1 = {}, s2 = {}, s3 = {};
#pragma unroll
        for (int kk = 0; kk < 2; ++kk) {
            const int col = 32 * kk + 16 * hi;
            s0 = WMMA_F16(aq[kk].v, load_bfrag(kcur,  0 + n, col), s0);
            s1 = WMMA_F16(aq[kk].v, load_bfrag(kcur, 16 + n, col), s1);
            s2 = WMMA_F16(aq[kk].v, load_bfrag(kcur, 32 + n, col), s2);
            s3 = WMMA_F16(aq[kk].v, load_bfrag(kcur, 48 + n, col), s3);
        }

        // ---- online softmax: row max via DPP16, exp, rescale; P -> LDS (f16)
#pragma unroll
        for (int g = 0; g < 8; ++g) {
            float t = fmaxf(fmaxf(s0[g], s1[g]), fmaxf(s2[g], s3[g]));
            t = rowmax16(t);
            float mnew  = fmaxf(m_r[g], t);
            float alpha = __builtin_amdgcn_exp2f(m_r[g] - mnew);
            m_r[g] = mnew;

            float p0 = __builtin_amdgcn_exp2f(s0[g] - mnew);
            float p1 = __builtin_amdgcn_exp2f(s1[g] - mnew);
            float p2 = __builtin_amdgcn_exp2f(s2[g] - mnew);
            float p3 = __builtin_amdgcn_exp2f(s3[g] - mnew);

            const int r = g + 8 * hi;
            myp[r * 64 +  0 + n] = (_Float16)p0;
            myp[r * 64 + 16 + n] = (_Float16)p1;
            myp[r * 64 + 32 + n] = (_Float16)p2;
            myp[r * 64 + 48 + n] = (_Float16)p3;

            o0[g] *= alpha; o1[g] *= alpha; o2[g] *= alpha; o3[g] *= alpha;
            lacc[g] *= alpha;
        }

        // ---- O += P * V ; l += P * ones   (row sums on the matrix core)
#pragma unroll
        for (int kk = 0; kk < 2; ++kk) {
            Frag16 ap;
            const _Float16* pr = myp + n * 64 + 32 * kk;
            ap.q[0] = *(const uint4*)(pr + 16 * hi);
            ap.q[1] = *(const uint4*)(pr + 8 + 16 * hi);
            const int col = 32 * kk + 16 * hi;
            o0   = WMMA_F16(ap.v, load_bfrag(vcur,  0 + n, col), o0);
            o1   = WMMA_F16(ap.v, load_bfrag(vcur, 16 + n, col), o1);
            o2   = WMMA_F16(ap.v, load_bfrag(vcur, 32 + n, col), o2);
            o3   = WMMA_F16(ap.v, load_bfrag(vcur, 48 + n, col), o3);
            lacc = WMMA_F16(ap.v, vones, lacc);
        }

        wait_async0();     // next block's tiles landed in LDS
        __syncthreads();   // all waves done reading current tiles
    }

    // ---- normalize + write f16 attention output (B,S,D)
#pragma unroll
    for (int g = 0; g < 8; ++g) {
        float inv = 1.0f / lacc[g];
        int r = qrow0 + g + 8 * hi;
        _Float16* orow = xh + base + (long)r * rowStride;
        orow[ 0 + n] = (_Float16)(o0[g] * inv);
        orow[16 + n] = (_Float16)(o1[g] * inv);
        orow[32 + n] = (_Float16)(o2[g] * inv);
        orow[48 + n] = (_Float16)(o3[g] * inv);
    }
}

// ---------------------------------------------------------------- projection GEMM
// out = x @ W^T + b, PReLU.  M=4096, N=1024, K=1024.
__global__ __launch_bounds__(256)
void proj_kernel(const _Float16* __restrict__ xh, const _Float16* __restrict__ wh,
                 const float* __restrict__ bias, const float* __restrict__ prelu,
                 float* __restrict__ out) {
    __shared__ _Float16 abuf[2][128 * 64];  // x block [row][k]
    __shared__ _Float16 bbuf[2][64 * 64];   // W block [j][k]

    const int tid  = threadIdx.x;
    const int lane = tid & 31;
    const int wave = tid >> 5;
    const int n    = lane & 15;
    const int hi   = lane >> 4;

    const int j0 = blockIdx.x * 64;
    const int m0 = blockIdx.y * 128;

    auto stage = [&](int kc, int bufi) {
#pragma unroll
        for (int u = 0; u < 4; ++u) {
            int c = tid + u * 256;
            async_copy16(&abuf[bufi][c * 8],
                         xh + (long)(m0 + (c >> 3)) * DMODEL + kc * 64 + (c & 7) * 8);
        }
#pragma unroll
        for (int u = 0; u < 2; ++u) {
            int c = tid + u * 256;
            async_copy16(&bbuf[bufi][c * 8],
                         wh + (long)(j0 + (c >> 3)) * DMODEL + kc * 64 + (c & 7) * 8);
        }
    };

    v8f acc0 = {}, acc1 = {}, acc2 = {}, acc3 = {};

    const int NK = DMODEL / 64;
    stage(0, 0);
    wait_async0();
    __syncthreads();

    for (int kc = 0; kc < NK; ++kc) {
        const int cur = kc & 1;
        if (kc + 1 < NK) stage(kc + 1, cur ^ 1);

        const _Float16* acur = abuf[cur];
        const _Float16* bcur = bbuf[cur];
#pragma unroll
        for (int kk = 0; kk < 2; ++kk) {
            Frag16 a;
            const _Float16* ar = acur + (wave * 16 + n) * 64 + 32 * kk;
            a.q[0] = *(const uint4*)(ar + 16 * hi);
            a.q[1] = *(const uint4*)(ar + 8 + 16 * hi);
            const int col = 32 * kk + 16 * hi;
            acc0 = WMMA_F16(a.v, load_bfrag(bcur,  0 + n, col), acc0);
            acc1 = WMMA_F16(a.v, load_bfrag(bcur, 16 + n, col), acc1);
            acc2 = WMMA_F16(a.v, load_bfrag(bcur, 32 + n, col), acc2);
            acc3 = WMMA_F16(a.v, load_bfrag(bcur, 48 + n, col), acc3);
        }
        wait_async0();
        __syncthreads();
    }

    const float ap = prelu[0];
#pragma unroll
    for (int g = 0; g < 8; ++g) {
        int r = m0 + wave * 16 + g + 8 * hi;
        float* orow = out + (long)r * DMODEL + j0;
        float v0 = acc0[g] + bias[j0 +  0 + n];
        float v1 = acc1[g] + bias[j0 + 16 + n];
        float v2 = acc2[g] + bias[j0 + 32 + n];
        float v3 = acc3[g] + bias[j0 + 48 + n];
        orow[ 0 + n] = v0 >= 0.0f ? v0 : ap * v0;
        orow[16 + n] = v1 >= 0.0f ? v1 : ap * v1;
        orow[32 + n] = v2 >= 0.0f ? v2 : ap * v2;
        orow[48 + n] = v3 >= 0.0f ? v3 : ap * v3;
    }
}

// ---------------------------------------------------------------- launch
extern "C" void kernel_launch(void* const* d_in, const int* in_sizes, int n_in,
                              void* d_out, int out_size, void* d_ws, size_t ws_size,
                              hipStream_t stream) {
    const float* q     = (const float*)d_in[0];
    const float* k     = (const float*)d_in[1];
    const float* v     = (const float*)d_in[2];
    const float* W     = (const float*)d_in[3];
    const float* bias  = (const float*)d_in[4];
    const float* prelu = (const float*)d_in[5];

    const long NQ = (long)BATCH * SLEN * DMODEL;   // 4,194,304
    _Float16* qh = (_Float16*)d_ws;
    _Float16* kh = qh + NQ;
    _Float16* vt = kh + NQ;     // V transposed: (B,H,hd,S)
    _Float16* xh = vt + NQ;
    _Float16* wh = xh + NQ;     // + 1M halves

    const float LOG2E = 1.44269504088896340736f;
    convert_kernel<<<2048, 256, 0, stream>>>(q, qh, (int)NQ, 0.125f * LOG2E);
    convert_kernel<<<2048, 256, 0, stream>>>(k, kh, (int)NQ, 1.0f);
    convert_kernel<<<1024, 256, 0, stream>>>(W, wh, DMODEL * DMODEL, 1.0f);
    convert_v_transpose<<<dim3(32, 32), 256, 0, stream>>>(v, vt);

    attn_kernel<<<dim3(16, 32), 256, 0, stream>>>(qh, kh, vt, xh);
    proj_kernel<<<dim3(16, 32), 256, 0, stream>>>(xh, wh, bias, prelu, (float*)d_out);
}
